// CompGCN_ConvE_82978768159424
// MI455X (gfx1250) — matrix operand order repaired and checked by reading, not compiled, not run on hardware
//
#include <hip/hip_runtime.h>
#include <math.h>

// ---------------- problem constants ----------------
#define N_ENT   50000
#define NREL2   400        // 2*N_REL
#define D0c     100
#define Dc      200
#define EHALF   250000     // E per direction
#define BQ      512
#define NFILT   200
#define OUTHW   196        // 14*14
#define FLATC   39200      // 196*200
#define KCONV   49
#define KCPAD   52         // conv K padded to 16B-aligned rows
#define EPSBN   1e-5f
#define NPAD    256        // padded N for async-copied weight tiles
#define KPAD1   128        // D0c=100 -> ceil32 -> 128
#define KPAD2   224        // Dc=200 -> ceil32 -> 224

typedef __bf16 bf16_t;
typedef __attribute__((ext_vector_type(16))) __bf16 v16bf;
typedef __attribute__((ext_vector_type(8)))  __bf16 v8bf;
typedef __attribute__((ext_vector_type(4)))  __bf16 v4bf;
typedef __attribute__((ext_vector_type(8)))  float  v8f;
typedef __attribute__((ext_vector_type(4)))  float  v4f;

#define LDSTRIDE 40   // bf16 elems per LDS row: 80 bytes, 16B-aligned rows

static __device__ inline bf16_t f2bf(float f) {
  union { float f; unsigned u; } c; c.f = f;
  unsigned u = c.u;
  unsigned r = (u + 0x7FFFu + ((u >> 16) & 1u)) >> 16;   // round-to-nearest-even
  unsigned short s = (unsigned short)r;
  return __builtin_bit_cast(bf16_t, s);
}

static __device__ inline v8f wmma_bf16(v16bf a, v16bf b, v8f c) {
  return __builtin_amdgcn_wmma_f32_16x16x32_bf16(false, a, false, b, (short)0, c, false, false);
}

// A fragment: lane L: row m=base+(L&15), h=L>>4; elem j holds K=(j<8)?8h+j:16+8h+(j-8)
static __device__ inline v16bf load_a_frag(const bf16_t* As, int mBase, int lane) {
  int m = mBase + (lane & 15);
  int h = lane >> 4;
  const v8bf* lo = (const v8bf*)(As + m * LDSTRIDE + 8 * h);
  const v8bf* hi = (const v8bf*)(As + m * LDSTRIDE + 16 + 8 * h);
  return __builtin_shufflevector(lo[0], hi[0],
                                 0, 1, 2, 3, 4, 5, 6, 7, 8, 9, 10, 11, 12, 13, 14, 15);
}
// B fragment from LDS tile stored TRANSPOSED (Bt[n][kk]): elem j holds K=(j<8)?8h+j:16+8h+(j-8)
static __device__ inline v16bf load_b_frag(const bf16_t* Bt, int nBase, int lane) {
  int n = nBase + (lane & 15);
  int h = lane >> 4;
  const v8bf* lo = (const v8bf*)(Bt + n * LDSTRIDE + 8 * h);
  const v8bf* hi = (const v8bf*)(Bt + n * LDSTRIDE + 16 + 8 * h);
  return __builtin_shufflevector(lo[0], hi[0],
                                 0, 1, 2, 3, 4, 5, 6, 7, 8, 9, 10, 11, 12, 13, 14, 15);
}

template<int EPI>  // 0=store, 1=store+bias, 2=sigmoid(acc+bias)
static __device__ inline void store_epi(float* C, int ldc, const float* bias, int m, int n, float v) {
  if (EPI >= 1 && bias) v += bias[n];
  if (EPI == 2) v = 1.0f / (1.0f + expf(-v));
  C[(size_t)m * ldc + n] = v;
}

// ---------------- generic WMMA GEMM: C[M,N] = A[M,K] @ B ----------------
// 256 threads, macro tile 128x64, 8 waves each own 32x32 (2x2 fragments, 4 WMMA/K-step)
template<bool TRANSB, int EPI>
__global__ __launch_bounds__(256) void gemm_wmma(
    const float* __restrict__ A, int lda,
    const float* __restrict__ B, int ldb,
    float* __restrict__ C, int ldc,
    const float* __restrict__ bias,
    int M, int N, int K) {
  __shared__ __align__(16) bf16_t As[128 * LDSTRIDE];
  __shared__ __align__(16) bf16_t Bt[64 * LDSTRIDE];
  const int t = threadIdx.x, lane = t & 31, w = t >> 5;
  const int mr = w >> 1, nc = w & 1;
  const int m0 = blockIdx.y * 128, n0 = blockIdx.x * 64;
  v8f acc[2][2] = {{{}, {}}, {{}, {}}};
  for (int k0 = 0; k0 < K; k0 += 32) {
#pragma unroll
    for (int i = 0; i < 4; ++i) {          // A tile 128x32, 4 floats per thread-step
      int r = (t >> 3) + 32 * i, q = t & 7;
      int gm = m0 + r, gk = k0 + q * 4;
      v4bf o;
      if (gm < M && gk + 3 < K) {
        v4f av = *(const v4f*)(A + (size_t)gm * lda + gk);
#pragma unroll
        for (int j = 0; j < 4; ++j) o[j] = f2bf(av[j]);
      } else {
#pragma unroll
        for (int j = 0; j < 4; ++j)
          o[j] = f2bf((gm < M && gk + j < K) ? A[(size_t)gm * lda + gk + j] : 0.0f);
      }
      *(v4bf*)(As + r * LDSTRIDE + q * 4) = o;
    }
    if (TRANSB) {
#pragma unroll
      for (int i = 0; i < 2; ++i) {        // B tile: 64 n x 8 k-quads
        int id = i * 256 + t;
        int n = id >> 3, q = id & 7;
        int gn = n0 + n, gk = k0 + q * 4;
        v4bf o;
        if (gn < N && gk + 3 < K) {
          v4f bv = *(const v4f*)(B + (size_t)gn * ldb + gk);
#pragma unroll
          for (int j = 0; j < 4; ++j) o[j] = f2bf(bv[j]);
        } else {
#pragma unroll
          for (int j = 0; j < 4; ++j)
            o[j] = f2bf((gn < N && gk + j < K) ? B[(size_t)gn * ldb + gk + j] : 0.0f);
        }
        *(v4bf*)(Bt + n * LDSTRIDE + q * 4) = o;
      }
    } else {
#pragma unroll
      for (int i = 0; i < 2; ++i) {        // B tile: 32 kk x 16 n-quads (transpose into Bt)
        int id = i * 256 + t;
        int kk = id >> 4, nq = id & 15;
        int gk = k0 + kk, gn = n0 + nq * 4;
        if (gk < K && gn + 3 < N) {
          v4f bv = *(const v4f*)(B + (size_t)gk * ldb + gn);
#pragma unroll
          for (int j = 0; j < 4; ++j) Bt[(nq * 4 + j) * LDSTRIDE + kk] = f2bf(bv[j]);
        } else {
#pragma unroll
          for (int j = 0; j < 4; ++j)
            Bt[(nq * 4 + j) * LDSTRIDE + kk] =
                f2bf((gk < K && gn + j < N) ? B[(size_t)gk * ldb + gn + j] : 0.0f);
        }
      }
    }
    if (k0 + 32 < K) {                     // gfx1250 global_prefetch_b8
      int r = t >> 1, k = (t & 1) * 16;
      int gm = m0 + r, gk = k0 + 32 + k;
      if (gm < M && gk < K) __builtin_prefetch(&A[(size_t)gm * lda + gk], 0, 0);
    }
    __syncthreads();
    v16bf a0 = load_a_frag(As, mr * 32, lane);
    v16bf a1 = load_a_frag(As, mr * 32 + 16, lane);
    v16bf b0 = load_b_frag(Bt, nc * 32, lane);
    v16bf b1 = load_b_frag(Bt, nc * 32 + 16, lane);
    acc[0][0] = wmma_bf16(a0, b0, acc[0][0]);
    acc[0][1] = wmma_bf16(a0, b1, acc[0][1]);
    acc[1][0] = wmma_bf16(a1, b0, acc[1][0]);
    acc[1][1] = wmma_bf16(a1, b1, acc[1][1]);
    __syncthreads();
  }
  int h = lane >> 4;
#pragma unroll
  for (int mi = 0; mi < 2; ++mi) {
#pragma unroll
    for (int ni = 0; ni < 2; ++ni) {
      int gn = n0 + nc * 32 + ni * 16 + (lane & 15);
      if (gn >= N) continue;
#pragma unroll
      for (int i = 0; i < 8; ++i) {
        int gm = m0 + mr * 32 + mi * 16 + i + 8 * h;
        if (gm < M) store_epi<EPI>(C, ldc, bias, gm, gn, acc[mi][ni][i]);
      }
    }
  }
}

// ---------------- fused edge-message GEMM with scatter-add epilogue ----------------
// A[e,k] = X[col[e],k]*R[et[e],k]*nrm[e]; Out[row[e],:] += A @ W  (atomic)
// W pre-converted to bf16, transposed + zero-padded: Wt[NPAD][Kpad].
// B tile staged with gfx1250 GLOBAL_LOAD_ASYNC_TO_LDS_B128 (ASYNCcnt path).
__global__ __launch_bounds__(256) void edge_gemm_wmma(
    const float* __restrict__ X, int ldx,
    const float* __restrict__ R, int ldr,
    const int* __restrict__ colIdx, const int* __restrict__ rowIdx,
    const int* __restrict__ etIdx, const float* __restrict__ nrm,
    const bf16_t* __restrict__ Wt, int Kpad,
    float* __restrict__ Out, int M, int K, int N) {
  __shared__ __align__(16) bf16_t As[128 * LDSTRIDE];
  __shared__ __align__(16) bf16_t Bt[64 * LDSTRIDE];
  __shared__ int sCol[128], sRow[128], sRel[128];
  __shared__ float sNrm[128];
  const int t = threadIdx.x, lane = t & 31, w = t >> 5;
  const int mr = w >> 1, nc = w & 1;
  const int m0 = blockIdx.y * 128, n0 = blockIdx.x * 64;
  if (t < 128) {
    int e = m0 + t;
    int c = 0, rr = 0, rl = 0; float nm = 0.0f;
    if (e < M) {
      c  = colIdx ? colIdx[e] : e;
      rr = rowIdx ? rowIdx[e] : e;
      rl = etIdx  ? etIdx[e]  : 0;
      nm = nrm    ? nrm[e]    : 1.0f;
    }
    sCol[t] = c; sRow[t] = rr; sRel[t] = rl; sNrm[t] = nm;
  }
  __syncthreads();
  // hoist this thread's 4 source-row pointers out of the K loop
  const float* xrow[4]; const float* rrow[4]; float mynrm[4];
#pragma unroll
  for (int i = 0; i < 4; ++i) {
    int r = (t >> 3) + 32 * i;
    xrow[i]  = X + (size_t)sCol[r] * ldx;
    rrow[i]  = R + (size_t)sRel[r] * ldr;
    mynrm[i] = sNrm[r];
  }
  v8f acc[2][2] = {{{}, {}}, {{}, {}}};
  for (int k0 = 0; k0 < K; k0 += 32) {
    {  // async-copy 32x64 bf16 weight tile: 256 lanes x 16B, no transpose needed
      int n = t >> 2, kseg = t & 3;
      const bf16_t* gsrc = Wt + (size_t)(n0 + n) * Kpad + k0 + kseg * 8;
      unsigned ldsoff = (unsigned)(uintptr_t)(const void*)&Bt[n * LDSTRIDE + kseg * 8];
      asm volatile("global_load_async_to_lds_b128 %0, %1, off"
                   :: "v"(ldsoff), "v"((unsigned long long)(uintptr_t)gsrc)
                   : "memory");
    }
#pragma unroll
    for (int i = 0; i < 4; ++i) {          // composed A tile: vectorized gather*rel*norm
      int r = (t >> 3) + 32 * i, q = t & 7;
      int gk = k0 + q * 4;
      v4bf o;
      if (gk + 3 < K) {
        v4f xv = *(const v4f*)(xrow[i] + gk);
        v4f rv = *(const v4f*)(rrow[i] + gk);
#pragma unroll
        for (int j = 0; j < 4; ++j) o[j] = f2bf(xv[j] * rv[j] * mynrm[i]);
      } else {
#pragma unroll
        for (int j = 0; j < 4; ++j)
          o[j] = f2bf((gk + j < K) ? xrow[i][gk + j] * rrow[i][gk + j] * mynrm[i] : 0.0f);
      }
      *(v4bf*)(As + r * LDSTRIDE + q * 4) = o;
    }
    if (k0 + 32 < K)
      __builtin_prefetch(xrow[0] + k0 + 32 + (t & 7) * 4, 0, 0);
    asm volatile("s_wait_asynccnt 0" ::: "memory");
    __syncthreads();
    v16bf a0 = load_a_frag(As, mr * 32, lane);
    v16bf a1 = load_a_frag(As, mr * 32 + 16, lane);
    v16bf b0 = load_b_frag(Bt, nc * 32, lane);
    v16bf b1 = load_b_frag(Bt, nc * 32 + 16, lane);
    acc[0][0] = wmma_bf16(a0, b0, acc[0][0]);
    acc[0][1] = wmma_bf16(a0, b1, acc[0][1]);
    acc[1][0] = wmma_bf16(a1, b0, acc[1][0]);
    acc[1][1] = wmma_bf16(a1, b1, acc[1][1]);
    __syncthreads();
  }
  int h = lane >> 4;
#pragma unroll
  for (int mi = 0; mi < 2; ++mi) {
#pragma unroll
    for (int ni = 0; ni < 2; ++ni) {
      int gn = n0 + nc * 32 + ni * 16 + (lane & 15);
      if (gn >= N) continue;
#pragma unroll
      for (int i = 0; i < 8; ++i) {
        int ml = mr * 32 + mi * 16 + i + 8 * h;
        if (m0 + ml < M)
          atomicAdd(&Out[(size_t)sRow[ml] * N + gn], acc[mi][ni][i]);
      }
    }
  }
}

// ---------------- fused FC GEMM: A = relu(BN1(conv_out)) with [b, f*196+pos] remap ----------------
__global__ __launch_bounds__(256) void fc_gemm_wmma(
    const float* __restrict__ Conv,   // [B*196, 200]
    const float* __restrict__ scale, const float* __restrict__ shift,  // BN1 per filter
    const float* __restrict__ B, const float* __restrict__ bias,
    float* __restrict__ C, int M, int N, int K) {
  __shared__ __align__(16) bf16_t As[128 * LDSTRIDE];
  __shared__ __align__(16) bf16_t Bt[64 * LDSTRIDE];
  const int t = threadIdx.x, lane = t & 31, w = t >> 5;
  const int mr = w >> 1, nc = w & 1;
  const int m0 = blockIdx.y * 128, n0 = blockIdx.x * 64;
  v8f acc[2][2] = {{{}, {}}, {{}, {}}};
  for (int k0 = 0; k0 < K; k0 += 32) {
#pragma unroll
    for (int i = 0; i < 16; ++i) {         // A is a strided gather (flatten remap): scalar
      int id = i * 256 + t;
      int r = id >> 5, k = id & 31;
      int gm = m0 + r, gk = k0 + k;
      float v = 0.0f;
      if (gm < M && gk < K) {
        int f = gk / OUTHW, pos = gk - f * OUTHW;
        float raw = Conv[((size_t)gm * OUTHW + pos) * NFILT + f];
        v = fmaxf(raw * scale[f] + shift[f], 0.0f);
      }
      As[r * LDSTRIDE + k] = f2bf(v);
    }
#pragma unroll
    for (int i = 0; i < 2; ++i) {          // B tile: 32 kk x 16 n-quads, vectorized
      int id = i * 256 + t;
      int kk = id >> 4, nq = id & 15;
      int gk = k0 + kk, gn = n0 + nq * 4;
      if (gk < K && gn + 3 < N) {
        v4f bv = *(const v4f*)(B + (size_t)gk * N + gn);
#pragma unroll
        for (int j = 0; j < 4; ++j) Bt[(nq * 4 + j) * LDSTRIDE + kk] = f2bf(bv[j]);
      } else {
#pragma unroll
        for (int j = 0; j < 4; ++j)
          Bt[(nq * 4 + j) * LDSTRIDE + kk] =
              f2bf((gk < K && gn + j < N) ? B[(size_t)gk * N + gn + j] : 0.0f);
      }
    }
    if (k0 + 32 < K) {
      int kk = t >> 6, n = (t & 63);
      int gk = k0 + 32 + kk, gn = n0 + n;
      if (gk < K && gn < N) __builtin_prefetch(&B[(size_t)gk * N + gn], 0, 0);
    }
    __syncthreads();
    v16bf a0 = load_a_frag(As, mr * 32, lane);
    v16bf a1 = load_a_frag(As, mr * 32 + 16, lane);
    v16bf b0 = load_b_frag(Bt, nc * 32, lane);
    v16bf b1 = load_b_frag(Bt, nc * 32 + 16, lane);
    acc[0][0] = wmma_bf16(a0, b0, acc[0][0]);
    acc[0][1] = wmma_bf16(a0, b1, acc[0][1]);
    acc[1][0] = wmma_bf16(a1, b0, acc[1][0]);
    acc[1][1] = wmma_bf16(a1, b1, acc[1][1]);
    __syncthreads();
  }
  int h = lane >> 4;
#pragma unroll
  for (int mi = 0; mi < 2; ++mi) {
#pragma unroll
    for (int ni = 0; ni < 2; ++ni) {
      int gn = n0 + nc * 32 + ni * 16 + (lane & 15);
      if (gn >= N) continue;
#pragma unroll
      for (int i = 0; i < 8; ++i) {
        int gm = m0 + mr * 32 + mi * 16 + i + 8 * h;
        if (gm < M) C[(size_t)gm * N + gn] = acc[mi][ni][i] + bias[gn];
      }
    }
  }
}

// ---------------- small helper kernels ----------------
__global__ void fill_k(float* p, float v, int n) {
  int i = blockIdx.x * blockDim.x + threadIdx.x;
  if (i < n) p[i] = v;
}

// W[K,N] f32 -> Wt[NPAD,Kpad] bf16 (transposed, zero-padded)
__global__ void wt_bf16_k(const float* __restrict__ W, bf16_t* Wt, int K, int N, int Kpad) {
  int i = blockIdx.x * blockDim.x + threadIdx.x;
  if (i < NPAD * Kpad) {
    int n = i / Kpad, k = i - n * Kpad;
    float v = (n < N && k < K) ? W[(size_t)k * N + n] : 0.0f;
    Wt[i] = f2bf(v);
  }
}

// conv_w[200][49] -> padded [200][52]
__global__ void convw_pad_k(const float* __restrict__ W, float* Wp) {
  int i = blockIdx.x * blockDim.x + threadIdx.x;
  if (i < NFILT * KCPAD) {
    int f = i / KCPAD, k = i - f * KCPAD;
    Wp[i] = (k < KCONV) ? W[(size_t)f * KCONV + k] : 0.0f;
  }
}

__global__ void degree_k(const int* __restrict__ ei0, float* degIn, float* degOut, int E) {
  int e = blockIdx.x * blockDim.x + threadIdx.x;
  if (e < 2 * E) {
    float* d = (e < E) ? degIn : degOut;
    atomicAdd(&d[ei0[e]], 1.0f);
  }
}

__global__ void rsqrt_k(float* p, int n) {
  int i = blockIdx.x * blockDim.x + threadIdx.x;
  if (i < n) { float v = p[i]; p[i] = (v > 0.0f) ? rsqrtf(v) : 0.0f; }
}

__global__ void nrm_k(const int* __restrict__ ei0, const int* __restrict__ ei1,
                      const float* __restrict__ dinv, float* nrm, int E) {
  int e = blockIdx.x * blockDim.x + threadIdx.x;
  if (e < 2 * E) {
    const float* dv = (e < E) ? dinv : (dinv + N_ENT);
    nrm[e] = dv[ei0[e]] * dv[ei1[e]];
  }
}

// per-column sum / sumsq over row-major [R,C], C<=256
__global__ void colstats_k(const float* __restrict__ X, int R, int C, float* s1, float* s2) {
  int c = threadIdx.x;
  if (c >= C) return;
  float a = 0.0f, b = 0.0f;
  for (int r = blockIdx.x; r < R; r += gridDim.x) {
    float v = X[(size_t)r * C + c];
    a += v; b += v * v;
  }
  atomicAdd(&s1[c], a);
  atomicAdd(&s2[c], b);
}

__global__ void fullstats_k(const float* __restrict__ X, int n, float* s1, float* s2) {
  __shared__ float sa[256], sb[256];
  float a = 0.0f, b = 0.0f;
  for (int i = blockIdx.x * blockDim.x + threadIdx.x; i < n; i += gridDim.x * blockDim.x) {
    float v = X[i]; a += v; b += v * v;
  }
  sa[threadIdx.x] = a; sb[threadIdx.x] = b;
  __syncthreads();
  for (int s = 128; s > 0; s >>= 1) {
    if (threadIdx.x < s) { sa[threadIdx.x] += sa[threadIdx.x + s]; sb[threadIdx.x] += sb[threadIdx.x + s]; }
    __syncthreads();
  }
  if (threadIdx.x == 0) { atomicAdd(s1, sa[0]); atomicAdd(s2, sb[0]); }
}

// fold BN stats + optional premul (val = raw*premul) into scale/shift
__global__ void bn_finalize_k(const float* s1, const float* s2, const float* g, const float* b,
                              float* scale, float* shift, int C, float invR, float premul) {
  int c = threadIdx.x;
  if (c < C) {
    float mean = premul * s1[c] * invR;
    float var  = premul * premul * s2[c] * invR - mean * mean;
    float tt   = g[c] * rsqrtf(var + EPSBN);
    scale[c] = premul * tt;
    shift[c] = b[c] - mean * tt;
  }
}

__global__ void bn_tanh_k(float* x, const float* scale, const float* shift, int n, int C) {
  int i = blockIdx.x * blockDim.x + threadIdx.x;
  if (i < n) { int c = i % C; x[i] = tanhf(x[i] * scale[c] + shift[c]); }
}

__global__ void bn_relu_k(const float* __restrict__ in, float* out,
                          const float* scale, const float* shift, int n, int C) {
  int i = blockIdx.x * blockDim.x + threadIdx.x;
  if (i < n) { int c = i % C; out[i] = fmaxf(in[i] * scale[c] + shift[c], 0.0f); }
}

__global__ void concat_rows_k(float* dst, const float* a, int rowsA, const float* b, int rowsB, int C) {
  int i = blockIdx.x * blockDim.x + threadIdx.x;
  int total = (rowsA + rowsB) * C;
  if (i < total) dst[i] = (i < rowsA * C) ? a[i] : b[i - rowsA * C];
}

// [B,2,D]->transpose->[B,D,2]->reshape[B,20,20]; flat p = d*2+c
__global__ void stack_k(const int* __restrict__ sub, const int* __restrict__ rel,
                        const float* __restrict__ x, const float* __restrict__ r, float* stk) {
  int i = blockIdx.x * blockDim.x + threadIdx.x;
  if (i < BQ * 400) {
    int b = i / 400, p = i % 400;
    int d = p >> 1, c = p & 1;
    stk[i] = c ? r[(size_t)rel[b] * Dc + d] : x[(size_t)sub[b] * Dc + d];
  }
}

// im2col with BN0 folded in; writes KCPAD-wide zero-padded rows
__global__ void im2col_k(const float* __restrict__ stk, const float* scale, const float* shift,
                         float* patch) {
  int i = blockIdx.x * blockDim.x + threadIdx.x;
  if (i < BQ * OUTHW * KCPAD) {
    int row = i / KCPAD, tt = i - row * KCPAD;
    float v = 0.0f;
    if (tt < KCONV) {
      int ky = tt / 7, kx = tt - ky * 7;
      int b = row / OUTHW, pos = row - b * OUTHW;
      int oy = pos / 14, ox = pos - oy * 14;
      v = stk[(size_t)b * 400 + (oy + ky) * 20 + (ox + kx)] * scale[0] + shift[0];
    }
    patch[i] = v;
  }
}

// ---------------- host driver ----------------
static inline int ceildiv(int a, int b) { return (a + b - 1) / b; }

extern "C" void kernel_launch(void* const* d_in, const int* in_sizes, int n_in,
                              void* d_out, int out_size, void* d_ws, size_t ws_size,
                              hipStream_t stream) {
  (void)in_sizes; (void)n_in; (void)out_size; (void)ws_size;
  const int*   sub        = (const int*)d_in[0];
  const int*   rel        = (const int*)d_in[1];
  const int*   ei0        = (const int*)d_in[2];
  const int*   ei1        = ei0 + 2 * EHALF;
  const int*   et         = (const int*)d_in[3];
  const float* init_embed = (const float*)d_in[4];
  const float* init_rel   = (const float*)d_in[5];
  const float* w_in1      = (const float*)d_in[6];
  const float* w_out1     = (const float*)d_in[7];
  const float* w_loop1    = (const float*)d_in[8];
  const float* w_rel1     = (const float*)d_in[9];
  const float* loop_rel1  = (const float*)d_in[10];
  const float* bn_g1      = (const float*)d_in[11];
  const float* bn_b1      = (const float*)d_in[12];
  const float* w_in2      = (const float*)d_in[13];
  const float* w_out2     = (const float*)d_in[14];
  const float* w_loop2    = (const float*)d_in[15];
  const float* w_rel2     = (const float*)d_in[16];
  const float* loop_rel2  = (const float*)d_in[17];
  const float* bn_g2      = (const float*)d_in[18];
  const float* bn_b2      = (const float*)d_in[19];
  const float* conv_w     = (const float*)d_in[20];
  const float* bn0_g      = (const float*)d_in[21];
  const float* bn0_b      = (const float*)d_in[22];
  const float* bn1_g      = (const float*)d_in[23];
  const float* bn1_b      = (const float*)d_in[24];
  const float* bn2_g      = (const float*)d_in[25];
  const float* bn2_b      = (const float*)d_in[26];
  const float* fc_w       = (const float*)d_in[27];
  const float* fc_b       = (const float*)d_in[28];
  const float* ent_bias   = (const float*)d_in[29];
  float* out = (float*)d_out;

  // bump allocator over workspace (~188 MB total)
  char* ws = (char*)d_ws;
  size_t off = 0;
  auto allocB = [&](size_t bytes) {
    void* p = (void*)(ws + off);
    off += ((bytes + 255) / 256) * 256;
    return p;
  };
  auto alloc = [&](size_t elems) { return (float*)allocB(elems * sizeof(float)); };
  float* dinv  = alloc(2 * N_ENT);                  // deg -> rsqrt in place (in | out)
  float* nrm   = alloc(2 * EHALF);
  float* x1    = alloc((size_t)N_ENT * Dc);
  float* x2    = alloc((size_t)N_ENT * Dc);
  float* rcat1 = alloc(401 * D0c);
  float* r1    = alloc(401 * Dc);
  float* rcat2 = alloc(401 * Dc);
  float* r2    = alloc(401 * Dc);
  float* stats = alloc(512);                        // s1 = stats, s2 = stats+256
  float* sscl  = alloc(512);                        // scale = sscl, shift = sscl+256
  float* stk   = alloc((size_t)BQ * 400);
  float* patch = alloc((size_t)BQ * OUTHW * KCPAD);
  float* convp = alloc((size_t)NFILT * KCPAD);
  float* conv  = alloc((size_t)BQ * OUTHW * NFILT);
  float* fco   = alloc((size_t)BQ * Dc);
  float* h2    = alloc((size_t)BQ * Dc);
  bf16_t* wt_in1   = (bf16_t*)allocB((size_t)NPAD * KPAD1 * 2);
  bf16_t* wt_out1  = (bf16_t*)allocB((size_t)NPAD * KPAD1 * 2);
  bf16_t* wt_loop1 = (bf16_t*)allocB((size_t)NPAD * KPAD1 * 2);
  bf16_t* wt_in2   = (bf16_t*)allocB((size_t)NPAD * KPAD2 * 2);
  bf16_t* wt_out2  = (bf16_t*)allocB((size_t)NPAD * KPAD2 * 2);
  bf16_t* wt_loop2 = (bf16_t*)allocB((size_t)NPAD * KPAD2 * 2);
  float* s1 = stats, *s2 = stats + 256;
  float* scl = sscl, *shf = sscl + 256;

  const int TB = 256;
  // ---- pre-convert edge-GEMM weights to transposed, padded bf16 (async-copy source) ----
  wt_bf16_k<<<ceildiv(NPAD * KPAD1, TB), TB, 0, stream>>>(w_in1,   wt_in1,   D0c, Dc, KPAD1);
  wt_bf16_k<<<ceildiv(NPAD * KPAD1, TB), TB, 0, stream>>>(w_out1,  wt_out1,  D0c, Dc, KPAD1);
  wt_bf16_k<<<ceildiv(NPAD * KPAD1, TB), TB, 0, stream>>>(w_loop1, wt_loop1, D0c, Dc, KPAD1);
  wt_bf16_k<<<ceildiv(NPAD * KPAD2, TB), TB, 0, stream>>>(w_in2,   wt_in2,   Dc,  Dc, KPAD2);
  wt_bf16_k<<<ceildiv(NPAD * KPAD2, TB), TB, 0, stream>>>(w_out2,  wt_out2,  Dc,  Dc, KPAD2);
  wt_bf16_k<<<ceildiv(NPAD * KPAD2, TB), TB, 0, stream>>>(w_loop2, wt_loop2, Dc,  Dc, KPAD2);

  // ---- degree / symmetric norm (shared by both layers) ----
  fill_k<<<ceildiv(2 * N_ENT, TB), TB, 0, stream>>>(dinv, 0.0f, 2 * N_ENT);
  degree_k<<<ceildiv(2 * EHALF, TB), TB, 0, stream>>>(ei0, dinv, dinv + N_ENT, EHALF);
  rsqrt_k<<<ceildiv(2 * N_ENT, TB), TB, 0, stream>>>(dinv, 2 * N_ENT);
  nrm_k<<<ceildiv(2 * EHALF, TB), TB, 0, stream>>>(ei0, ei1, dinv, nrm, EHALF);

  // ---- CompGCN layer 1 (Din=100 -> Dout=200) ----
  fill_k<<<ceildiv(N_ENT * Dc, TB), TB, 0, stream>>>(x1, 0.0f, N_ENT * Dc);
  {
    dim3 g(ceildiv(Dc, 64), ceildiv(EHALF, 128));
    edge_gemm_wmma<<<g, TB, 0, stream>>>(init_embed, D0c, init_rel, D0c,
        ei1, ei0, et, nrm, wt_in1, KPAD1, x1, EHALF, D0c, Dc);
    edge_gemm_wmma<<<g, TB, 0, stream>>>(init_embed, D0c, init_rel, D0c,
        ei1 + EHALF, ei0 + EHALF, et + EHALF, nrm + EHALF, wt_out1, KPAD1, x1, EHALF, D0c, Dc);
    dim3 gl(ceildiv(Dc, 64), ceildiv(N_ENT, 128));
    edge_gemm_wmma<<<gl, TB, 0, stream>>>(init_embed, D0c, loop_rel1, D0c,
        nullptr, nullptr, nullptr, nullptr, wt_loop1, KPAD1, x1, N_ENT, D0c, Dc);
  }
  fill_k<<<2, TB, 0, stream>>>(stats, 0.0f, 512);
  colstats_k<<<1024, TB, 0, stream>>>(x1, N_ENT, Dc, s1, s2);
  bn_finalize_k<<<1, TB, 0, stream>>>(s1, s2, bn_g1, bn_b1, scl, shf, Dc, 1.0f / N_ENT, 1.0f / 3.0f);
  bn_tanh_k<<<ceildiv(N_ENT * Dc, TB), TB, 0, stream>>>(x1, scl, shf, N_ENT * Dc, Dc);
  concat_rows_k<<<ceildiv(401 * D0c, TB), TB, 0, stream>>>(rcat1, init_rel, NREL2, loop_rel1, 1, D0c);
  {
    dim3 g(ceildiv(Dc, 64), ceildiv(401, 128));
    gemm_wmma<false, 0><<<g, TB, 0, stream>>>(rcat1, D0c, w_rel1, Dc, r1, Dc, nullptr, 401, Dc, D0c);
  }

  // ---- CompGCN layer 2 (Din=200 -> Dout=200) ----
  fill_k<<<ceildiv(N_ENT * Dc, TB), TB, 0, stream>>>(x2, 0.0f, N_ENT * Dc);
  {
    dim3 g(ceildiv(Dc, 64), ceildiv(EHALF, 128));
    edge_gemm_wmma<<<g, TB, 0, stream>>>(x1, Dc, r1, Dc,
        ei1, ei0, et, nrm, wt_in2, KPAD2, x2, EHALF, Dc, Dc);
    edge_gemm_wmma<<<g, TB, 0, stream>>>(x1, Dc, r1, Dc,
        ei1 + EHALF, ei0 + EHALF, et + EHALF, nrm + EHALF, wt_out2, KPAD2, x2, EHALF, Dc, Dc);
    dim3 gl(ceildiv(Dc, 64), ceildiv(N_ENT, 128));
    edge_gemm_wmma<<<gl, TB, 0, stream>>>(x1, Dc, loop_rel2, Dc,
        nullptr, nullptr, nullptr, nullptr, wt_loop2, KPAD2, x2, N_ENT, Dc, Dc);
  }
  fill_k<<<2, TB, 0, stream>>>(stats, 0.0f, 512);
  colstats_k<<<1024, TB, 0, stream>>>(x2, N_ENT, Dc, s1, s2);
  bn_finalize_k<<<1, TB, 0, stream>>>(s1, s2, bn_g2, bn_b2, scl, shf, Dc, 1.0f / N_ENT, 1.0f / 3.0f);
  bn_tanh_k<<<ceildiv(N_ENT * Dc, TB), TB, 0, stream>>>(x2, scl, shf, N_ENT * Dc, Dc);
  concat_rows_k<<<ceildiv(401 * Dc, TB), TB, 0, stream>>>(rcat2, r1, NREL2, loop_rel2, 1, Dc);
  {
    dim3 g(ceildiv(Dc, 64), ceildiv(401, 128));
    gemm_wmma<false, 0><<<g, TB, 0, stream>>>(rcat2, Dc, w_rel2, Dc, r2, Dc, nullptr, 401, Dc, Dc);
  }

  // ---- ConvE head ----
  stack_k<<<ceildiv(BQ * 400, TB), TB, 0, stream>>>(sub, rel, x2, r2, stk);
  fill_k<<<2, TB, 0, stream>>>(stats, 0.0f, 512);
  fullstats_k<<<256, TB, 0, stream>>>(stk, BQ * 400, s1, s2);
  bn_finalize_k<<<1, TB, 0, stream>>>(s1, s2, bn0_g, bn0_b, scl, shf, 1, 1.0f / (BQ * 400), 1.0f);
  im2col_k<<<ceildiv(BQ * OUTHW * KCPAD, TB), TB, 0, stream>>>(stk, scl, shf, patch);
  convw_pad_k<<<ceildiv(NFILT * KCPAD, TB), TB, 0, stream>>>(conv_w, convp);
  {  // conv as implicit GEMM: [B*196,52] @ convp^T[52,200] (zero-padded K)
    dim3 g(ceildiv(NFILT, 64), ceildiv(BQ * OUTHW, 128));
    gemm_wmma<true, 0><<<g, TB, 0, stream>>>(patch, KCPAD, convp, KCPAD, conv, NFILT,
                                             nullptr, BQ * OUTHW, NFILT, KCPAD);
  }
  fill_k<<<2, TB, 0, stream>>>(stats, 0.0f, 512);
  colstats_k<<<1024, TB, 0, stream>>>(conv, BQ * OUTHW, NFILT, s1, s2);
  bn_finalize_k<<<1, TB, 0, stream>>>(s1, s2, bn1_g, bn1_b, scl, shf, NFILT, 1.0f / (BQ * OUTHW), 1.0f);
  {  // FC with fused BN1+relu+flatten-remap on A
    dim3 g(ceildiv(Dc, 64), ceildiv(BQ, 128));
    fc_gemm_wmma<<<g, TB, 0, stream>>>(conv, scl, shf, fc_w, fc_b, fco, BQ, Dc, FLATC);
  }
  fill_k<<<2, TB, 0, stream>>>(stats, 0.0f, 512);
  colstats_k<<<512, TB, 0, stream>>>(fco, BQ, Dc, s1, s2);
  bn_finalize_k<<<1, TB, 0, stream>>>(s1, s2, bn2_g, bn2_b, scl, shf, Dc, 1.0f / BQ, 1.0f);
  bn_relu_k<<<ceildiv(BQ * Dc, TB), TB, 0, stream>>>(fco, h2, scl, shf, BQ * Dc, Dc);
  {  // logits = sigmoid(h2 @ x2^T + ent_bias)
    dim3 g(ceildiv(N_ENT, 64), ceildiv(BQ, 128));
    gemm_wmma<true, 2><<<g, TB, 0, stream>>>(h2, Dc, x2, Dc, out, N_ENT,
                                             ent_bias, BQ, N_ENT, Dc);
  }
}
